// TargetAttention_57999238365389
// MI455X (gfx1250) — compile-verified, matrix operand order checked by reading, not run on hardware
//
#include <hip/hip_runtime.h>
#include <math.h>

// Problem constants from the reference
#define DD   128   // feature dim
#define HH   128   // scorer hidden
#define TT   200   // sequence length
#define TPAD 208   // padded to 13 * 16 WMMA tiles
#define ASTR 132   // padded LDS row stride (bf16 elems): 66 dwords -> conflict-free rows, 4B aligned
#define WSTR 132   // same padding for W_eff^T

typedef __attribute__((ext_vector_type(16))) __bf16 v16bf;
typedef __attribute__((ext_vector_type(8)))  float  v8f;

union Frag { unsigned int u[8]; v16bf v; };

__device__ __forceinline__ unsigned short f2bf(float f) {
    unsigned int u = __float_as_uint(f);
    u += 0x7FFFu + ((u >> 16) & 1u);   // round-to-nearest-even
    return (unsigned short)(u >> 16);
}
__device__ __forceinline__ float bf2f(unsigned short h) {
    return __uint_as_float(((unsigned int)h) << 16);
}

// K offset inside a 16-bit A 16x32 / B 32x16 fragment (packed pair index i, lane half)
__device__ __forceinline__ int koff(int i, int half) {
    return ((i < 4) ? (2 * i) : (16 + 2 * (i - 4))) + 8 * half;
}

__launch_bounds__(256)
__global__ void target_attn(const float* __restrict__ tout,
                            const float* __restrict__ cand,
                            const unsigned char* __restrict__ mask,
                            const float* __restrict__ W1,
                            const float* __restrict__ b1,
                            const float* __restrict__ aPtr,
                            const float* __restrict__ W2,
                            const float* __restrict__ b2Ptr,
                            float* __restrict__ out) {
    __shared__ __align__(16) unsigned short A_lds[TPAD * ASTR];  // t_out[b] as bf16: 54,912 B
    __shared__ __align__(16) unsigned short Wt_lds[HH * WSTR];   // W_eff^T  as bf16: 33,792 B
    __shared__ __align__(16) float cand_lds[DD];
    __shared__ __align__(16) float bb_lds[HH];
    __shared__ __align__(16) float w2_lds[HH];
    __shared__ __align__(16) float scores[TPAD];
    __shared__ __align__(16) float weights[TPAD];
    __shared__ __align__(16) float red[256];
    __shared__ __align__(16) float pp[2][DD];

    const int b   = blockIdx.x;
    const int tid = threadIdx.x;
    const float alpha = aPtr[0];
    const float b2    = b2Ptr[0];

    // ---- phase 1: stage candidate + W2 ------------------------------------
    if (tid < DD) {
        cand_lds[tid] = cand[(size_t)b * DD + tid];
        w2_lds[tid]   = W2[tid];
    }
    __syncthreads();

    // ---- phase 2: stage t_out[b] (bf16), build W_eff^T (bf16) + eff bias --
    const float* tb = tout + (size_t)b * TT * DD;
    for (int idx = tid; idx < TT * DD; idx += 256) {
        int t = idx >> 7, d = idx & 127;
        A_lds[t * ASTR + d] = f2bf(tb[idx]);
    }
    for (int idx = tid; idx < (TPAD - TT) * DD; idx += 256) {
        int t = TT + (idx >> 7), d = idx & 127;
        A_lds[t * ASTR + d] = 0;                       // zero pad rows
    }
    // W_eff[k][h] = W1a[k][h] + cand[k]*W1c[k][h]; stored transposed: Wt[h][k]
    for (int idx = tid; idx < DD * HH; idx += 256) {
        int k = idx >> 7, h = idx & 127;               // coalesced over h
        float w = W1[(size_t)k * HH + h] + cand_lds[k] * W1[(size_t)(2 * DD + k) * HH + h];
        Wt_lds[h * WSTR + k] = f2bf(w);
    }
    if (tid < HH) {
        // bb[h] = b1[h] + cand_b @ W1b[:,h]   (coalesced over h per k step)
        float s = b1[tid];
        for (int k = 0; k < DD; ++k)
            s += cand_lds[k] * W1[(size_t)(DD + k) * HH + tid];
        bb_lds[tid] = s;
    }
    __syncthreads();

    // ---- phase 3: H^T = W_eff^T[128x128] @ t_out^T[128x208] via WMMA ------
    // C tile = [16 h-rows x 16 t-cols]: N(=t) in lanes, M(=h) in VGPRs/halves,
    // so the W2-dot reduction is register-local + ONE shfl_xor(16). No atomics.
    const int lane = tid & 31;
    const int half = lane >> 4;
    const int l15  = lane & 15;
    const int wave = tid >> 5;

    for (int tt = wave; tt < TPAD / 16; tt += 8) {
        const int tbase = tt * 16;

        // B fragments: t_out^T 32x16 tiles (lane = t column, packed d pairs)
        Frag bfr[4];
#pragma unroll
        for (int ks = 0; ks < 4; ++ks) {
#pragma unroll
            for (int i = 0; i < 8; ++i) {
                int k = ks * 32 + koff(i, half);
                bfr[ks].u[i] = *(const unsigned int*)&A_lds[(tbase + l15) * ASTR + k];
            }
        }

        float acc = 0.f;
#pragma unroll
        for (int ht = 0; ht < 8; ++ht) {
            const int hb = ht * 16;
            v8f c = {0.f, 0.f, 0.f, 0.f, 0.f, 0.f, 0.f, 0.f};
#pragma unroll
            for (int ks = 0; ks < 4; ++ks) {
                Frag af;  // A fragment: W_eff^T 16x32 tile (lane = h row)
#pragma unroll
                for (int i = 0; i < 8; ++i) {
                    int k = ks * 32 + koff(i, half);
                    af.u[i] = *(const unsigned int*)&Wt_lds[(hb + l15) * WSTR + k];
                }
                c = __builtin_amdgcn_wmma_f32_16x16x32_bf16(
                        false, af.v, false, bfr[ks].v, (short)0, c, false, false);
            }
            // bias + PReLU + * W2, reduced locally over this tile's 8 M(h)-rows
#pragma unroll
            for (int r = 0; r < 8; ++r) {
                const int hidx = hb + r + 8 * half;
                float hv = c[r] + bb_lds[hidx];
                hv = (hv >= 0.f) ? hv : alpha * hv;
                acc = fmaf(hv, w2_lds[hidx], acc);
            }
        }
        acc += __shfl_xor(acc, 16);          // merge the two M-halves
        if (lane < 16)
            scores[tbase + l15] = acc + b2;  // disjoint per wave: plain store
    }
    __syncthreads();

    // ---- phase 4: masked softmax over T=200 -------------------------------
    float sv = -__builtin_inff();
    bool mk = false;
    if (tid < TT) {
        mk = mask[(size_t)b * TT + tid] != 0;
        sv = mk ? scores[tid] : -__builtin_inff();
    }
    red[tid] = sv;
    __syncthreads();
    for (int o = 128; o >= 1; o >>= 1) {
        if (tid < o) red[tid] = fmaxf(red[tid], red[tid + o]);
        __syncthreads();
    }
    const float mx = red[0];
    __syncthreads();

    float ev = 0.f;
    if (tid < TT) {
        ev = mk ? __expf(scores[tid] - mx) : 0.f;
        weights[tid] = ev;
    }
    red[tid] = ev;
    __syncthreads();
    for (int o = 128; o >= 1; o >>= 1) {
        if (tid < o) red[tid] += red[tid + o];
        __syncthreads();
    }
    const float inv = 1.0f / red[0];
    __syncthreads();

    // ---- phase 5: out[b,d] = sum_t w[t] * t_out[b,t,d] (from LDS bf16) ----
    const int part = tid >> 7;
    const int d    = tid & 127;
    float acc2 = 0.f;
    const int t0 = part * (TT / 2), t1 = t0 + (TT / 2);
    for (int t = t0; t < t1; ++t)
        acc2 += weights[t] * bf2f(A_lds[t * ASTR + d]);
    pp[part][d] = acc2;
    __syncthreads();
    if (tid < DD)
        out[(size_t)b * DD + tid] = (pp[0][tid] + pp[1][tid]) * inv;
}

extern "C" void kernel_launch(void* const* d_in, const int* in_sizes, int n_in,
                              void* d_out, int out_size, void* d_ws, size_t ws_size,
                              hipStream_t stream) {
    const float*         tout = (const float*)d_in[0];
    const float*         cand = (const float*)d_in[1];
    const unsigned char* mask = (const unsigned char*)d_in[2];
    const float*         W1   = (const float*)d_in[3];
    const float*         b1   = (const float*)d_in[4];
    const float*         a    = (const float*)d_in[5];
    const float*         W2   = (const float*)d_in[6];
    const float*         b2   = (const float*)d_in[7];
    float*               out  = (float*)d_out;

    const int B = in_sizes[1] / DD;   // candidate is [B, D]
    target_attn<<<B, 256, 0, stream>>>(tout, cand, mask, W1, b1, a, W2, b2, out);
}
